// SequenceMemoryUpdater_9423158247658
// MI455X (gfx1250) — matrix operand order repaired and checked by reading, not compiled, not run on hardware
//
#include <hip/hip_runtime.h>

// ---------------------------------------------------------------------------
// TGN memory updater for MI455X (gfx1250, wave32).
//  - bf16 WMMA (v_wmma_f32_16x16x32_bf16) for both GEMMs, f32 accumulate
//  - full memory clone is the bandwidth term (~1GB); GEMM is far below roofline
// ---------------------------------------------------------------------------

typedef __attribute__((ext_vector_type(16))) __bf16 v16bf;
typedef __attribute__((ext_vector_type(8)))  __bf16 v8bf;
typedef __attribute__((ext_vector_type(8)))  float  v8f;

#define MEM_DIM  128
#define MSG_DIM  256
#define GATE_DIM 384   // 3 * MEM_DIM

// LDS row strides (elements), padded to break bank-conflict periodicity
constexpr int XSTRIDE  = 272; // bf16, 256 cols + pad (row stride 544B, 16B aligned)
constexpr int HBSTRIDE = 144; // bf16, 128 cols + pad (288B)
constexpr int HFSTRIDE = 136; // f32,  128 cols + pad (544B)

static __device__ __forceinline__ v8f wmma_bf16(v16bf a, v16bf b, v8f c) {
  // (neg_a, A, neg_b, B, c_mod, C, reuse_a, reuse_b)
  return __builtin_amdgcn_wmma_f32_16x16x32_bf16(false, a, false, b, (short)0, c,
                                                 false, false);
}

__global__ void cvt_bf16_kernel(const float* __restrict__ src,
                                __bf16* __restrict__ dst, int n) {
  int i = blockIdx.x * blockDim.x + threadIdx.x;
  if (i < n) dst[i] = (__bf16)src[i];
}

__global__ void copy_f4_kernel(const float4* __restrict__ src,
                               float4* __restrict__ dst, long long n4) {
  long long i = (long long)blockIdx.x * blockDim.x + threadIdx.x;
  long long stride = (long long)gridDim.x * blockDim.x;
  for (; i < n4; i += stride) dst[i] = src[i];
}

__global__ __launch_bounds__(512)
void gru_wmma_scatter_kernel(const float* __restrict__ memory,
                             const int*   __restrict__ uid,
                             const float* __restrict__ msgs,
                             const float* __restrict__ ts,
                             const float* __restrict__ b_ih,
                             const float* __restrict__ b_hh,
                             const __bf16* __restrict__ Wihb,   // [384,256] bf16
                             const __bf16* __restrict__ Whhb,   // [384,128] bf16
                             float* __restrict__ outMem,
                             float* __restrict__ outLU,
                             int Bn) {
  __shared__ __bf16 sX [32 * XSTRIDE];   // X tile   (32 x 256) bf16
  __shared__ __bf16 sHb[32 * HBSTRIDE];  // H tile   (32 x 128) bf16 (for WMMA)
  __shared__ float  sHf[32 * HFSTRIDE];  // H tile   (32 x 128) f32  (for blend)

  const int tid  = threadIdx.x;
  const int base = blockIdx.x * 32;

  // last_update scatter (independent of the GEMM; clone already stream-ordered)
  if (tid < 32) {
    int g = base + tid;
    if (g < Bn) outLU[uid[g]] = ts[g];
  }

  // ---- cooperative stage: X tile, 32 rows x 64 float4 ----
  for (int i = tid; i < 32 * (MSG_DIM / 4); i += 512) {
    int r  = i >> 6;
    int c4 = i & 63;
    int g  = base + r; if (g >= Bn) g = Bn - 1;          // clamp (uniform EXEC)
    float4 v = reinterpret_cast<const float4*>(msgs + (size_t)g * MSG_DIM)[c4];
    __bf16* p = &sX[r * XSTRIDE + c4 * 4];
    p[0] = (__bf16)v.x; p[1] = (__bf16)v.y; p[2] = (__bf16)v.z; p[3] = (__bf16)v.w;
  }
  // ---- cooperative stage: gathered H tile, 32 rows x 32 float4 ----
  for (int i = tid; i < 32 * (MEM_DIM / 4); i += 512) {
    int r  = i >> 5;
    int c4 = i & 31;
    int g  = base + r; if (g >= Bn) g = Bn - 1;
    int node = uid[g];
    float4 v = reinterpret_cast<const float4*>(memory + (size_t)node * MEM_DIM)[c4];
    float* pf = &sHf[r * HFSTRIDE + c4 * 4];
    pf[0] = v.x; pf[1] = v.y; pf[2] = v.z; pf[3] = v.w;
    __bf16* pb = &sHb[r * HBSTRIDE + c4 * 4];
    pb[0] = (__bf16)v.x; pb[1] = (__bf16)v.y; pb[2] = (__bf16)v.z; pb[3] = (__bf16)v.w;
  }
  __syncthreads();

  // ---- per-wave 16x16 tile: wave w = (mh, t); rows [base+mh*16, +16),
  //      gate columns [t*16, t*16+16) of each of r/z/n gate blocks ----
  const int lane = tid & 31;
  const int w    = tid >> 5;
  const int half = lane >> 4;   // lane group (K/N split per ISA layout)
  const int nn   = lane & 15;   // A: M index; B: N index; C/D: N index
  const int mh   = w >> 3;      // which 16-row half of the 32-row block tile
  const int t    = w & 7;       // gate-column tile (0..7 over MEM_DIM=128)

  v8f accR  = {0.f,0.f,0.f,0.f,0.f,0.f,0.f,0.f};
  v8f accZ  = {0.f,0.f,0.f,0.f,0.f,0.f,0.f,0.f};
  v8f accIN = {0.f,0.f,0.f,0.f,0.f,0.f,0.f,0.f};
  v8f accHN = {0.f,0.f,0.f,0.f,0.f,0.f,0.f,0.f};

  const int arow = (mh * 16 + nn) * XSTRIDE;
  const int hrow = (mh * 16 + nn) * HBSTRIDE;

  // ---- Gi = X * W_ih^T  (K = 256, 8 k-steps of 32) ----
#pragma unroll
  for (int ks = 0; ks < 8; ++ks) {
    const int k0 = ks * 32 + half * 16;     // A-frag: K {k0..k0+15} contiguous
    v8bf c0 = *reinterpret_cast<const v8bf*>(&sX[arow + k0]);
    v8bf c1 = *reinterpret_cast<const v8bf*>(&sX[arow + k0 + 8]);
    v16bf a = __builtin_shufflevector(c0, c1, 0,1,2,3,4,5,6,7,8,9,10,11,12,13,14,15);
    const int kb = ks * 32 + half * 16;     // B-frag: contiguous K run per lane
    v16bf br = *reinterpret_cast<const v16bf*>(Wihb + (size_t)(      t*16 + nn) * MSG_DIM + kb);
    v16bf bz = *reinterpret_cast<const v16bf*>(Wihb + (size_t)(128 + t*16 + nn) * MSG_DIM + kb);
    v16bf bn = *reinterpret_cast<const v16bf*>(Wihb + (size_t)(256 + t*16 + nn) * MSG_DIM + kb);
    accR  = wmma_bf16(a, br, accR);
    accZ  = wmma_bf16(a, bz, accZ);
    accIN = wmma_bf16(a, bn, accIN);
  }

  // ---- Gh = H * W_hh^T  (K = 128, 4 k-steps); r/z accumulate in place,
  //      n-gate of Gh kept separate (it is scaled by r before tanh) ----
#pragma unroll
  for (int ks = 0; ks < 4; ++ks) {
    const int k0 = ks * 32 + half * 16;
    v8bf c0 = *reinterpret_cast<const v8bf*>(&sHb[hrow + k0]);
    v8bf c1 = *reinterpret_cast<const v8bf*>(&sHb[hrow + k0 + 8]);
    v16bf a = __builtin_shufflevector(c0, c1, 0,1,2,3,4,5,6,7,8,9,10,11,12,13,14,15);
    const int kb = ks * 32 + half * 16;
    v16bf br = *reinterpret_cast<const v16bf*>(Whhb + (size_t)(      t*16 + nn) * MEM_DIM + kb);
    v16bf bz = *reinterpret_cast<const v16bf*>(Whhb + (size_t)(128 + t*16 + nn) * MEM_DIM + kb);
    v16bf bn = *reinterpret_cast<const v16bf*>(Whhb + (size_t)(256 + t*16 + nn) * MEM_DIM + kb);
    accR  = wmma_bf16(a, br, accR);
    accZ  = wmma_bf16(a, bz, accZ);
    accHN = wmma_bf16(a, bn, accHN);
  }

  // ---- gate math + scatter (C/D layout: reg r, lane l -> M=r+8*half, N=l&15) ----
  const int col  = t * 16 + nn;
  const float bihr = b_ih[col], bihz = b_ih[128 + col], bihn = b_ih[256 + col];
  const float bhhr = b_hh[col], bhhz = b_hh[128 + col], bhhn = b_hh[256 + col];

#pragma unroll
  for (int r = 0; r < 8; ++r) {
    const int mloc = mh * 16 + half * 8 + r;   // row within 32-row block tile
    const int g    = base + mloc;
    const float xr  = accR[r]  + bihr + bhhr;
    const float xz  = accZ[r]  + bihz + bhhz;
    const float xin = accIN[r] + bihn;
    const float xhn = accHN[r] + bhhn;
    const float rg = 1.f / (1.f + __expf(-xr));
    const float zg = 1.f / (1.f + __expf(-xz));
    const float nx = xin + rg * xhn;
    const float e2 = __expf(-2.f * nx);
    const float ng = (1.f - e2) / (1.f + e2);  // tanh
    const float h  = sHf[mloc * HFSTRIDE + col];
    const float hnew = (1.f - zg) * ng + zg * h;
    if (g < Bn) {
      const int node = uid[g];
      outMem[(size_t)node * MEM_DIM + col] = hnew;
    }
  }
}

extern "C" void kernel_launch(void* const* d_in, const int* in_sizes, int n_in,
                              void* d_out, int out_size, void* d_ws, size_t ws_size,
                              hipStream_t stream) {
  const float* memory      = (const float*)d_in[0];
  const float* last_update = (const float*)d_in[1];
  const int*   uid         = (const int*)d_in[2];
  const float* msgs        = (const float*)d_in[3];
  const float* ts          = (const float*)d_in[4];
  const float* W_ih        = (const float*)d_in[5];
  const float* W_hh        = (const float*)d_in[6];
  const float* b_ih        = (const float*)d_in[7];
  const float* b_hh        = (const float*)d_in[8];

  const long long n_mem   = (long long)in_sizes[0];       // n_nodes * 128
  const int       n_nodes = (int)(n_mem / MEM_DIM);
  const int       Bn      = in_sizes[2];

  float* outMem = (float*)d_out;
  float* outLU  = outMem + (size_t)n_nodes * MEM_DIM;

  // workspace: bf16 copies of the weights (L2-resident, 288 KB)
  __bf16* Wihb = (__bf16*)d_ws;
  __bf16* Whhb = Wihb + (size_t)GATE_DIM * MSG_DIM;

  const int nWih = GATE_DIM * MSG_DIM;   // 98304
  const int nWhh = GATE_DIM * MEM_DIM;   // 49152
  cvt_bf16_kernel<<<(nWih + 255) / 256, 256, 0, stream>>>(W_ih, Wihb, nWih);
  cvt_bf16_kernel<<<(nWhh + 255) / 256, 256, 0, stream>>>(W_hh, Whhb, nWhh);

  // functional clone: memory -> out, last_update -> out tail (pure BW term)
  copy_f4_kernel<<<8192, 256, 0, stream>>>(
      (const float4*)memory, (float4*)outMem, n_mem / 4);
  copy_f4_kernel<<<256, 256, 0, stream>>>(
      (const float4*)last_update, (float4*)outLU, (long long)n_nodes / 4);

  // GRU + scatter (stream-ordered after the clone)
  const int nBlocks = (Bn + 31) / 32;
  gru_wmma_scatter_kernel<<<nBlocks, 512, 0, stream>>>(
      memory, uid, msgs, ts, b_ih, b_hh, Wihb, Whhb, outMem, outLU, Bn);
}
